// MultiHeadedAttention_9268539425204
// MI455X (gfx1250) — matrix operand order repaired and checked by reading, not compiled, not run on hardware
//
#include <hip/hip_runtime.h>
#include <hip/hip_bf16.h>

#define B_ 4
#define S_ 2048
#define D_ 1024
#define H_ 16
#define DK_ 64

typedef __attribute__((ext_vector_type(16))) __bf16 v16bf;
typedef __attribute__((ext_vector_type(8)))  float  v8f;
typedef unsigned int u32x4 __attribute__((ext_vector_type(4)));
typedef int          i32x8 __attribute__((ext_vector_type(8)));
typedef int          i32x4 __attribute__((ext_vector_type(4)));

union FragU { v16bf v; unsigned u[8]; };

__device__ __forceinline__ unsigned short f2bf(float f) {
  union { float f; unsigned u; } a; a.f = f;
  unsigned u = a.u;
  u += 0x7FFFu + ((u >> 16) & 1u);   // round-to-nearest-even
  return (unsigned short)(u >> 16);
}

// A-matrix 16x32 bf16 fragment (ISA 7.12.2): lane m=lane&15, hi=lane>>4,
// VGPR v holds K pair k = (v/4)*16 + hi*8 + (v%4)*2  -> contiguous b32 reads.
__device__ __forceinline__ v16bf frag_a(const unsigned short* p, int stride, int m, int hi) {
  FragU f;
#pragma unroll
  for (int v = 0; v < 8; ++v) {
    int k = ((v >> 2) << 4) + (hi << 3) + ((v & 3) << 1);
    f.u[v] = *(const unsigned*)(p + m * stride + k);
  }
  return f.v;
}

// B-matrix 32x16 bf16 fragment: lane n=lane&15 holds column n,
// VGPR v holds K pair k = (v/4)*16 + hi*8 + (v%4)*2 -> contiguous b32 reads of the column source row.
__device__ __forceinline__ v16bf frag_b(const unsigned short* p, int stride, int n, int hi) {
  FragU f;
#pragma unroll
  for (int v = 0; v < 8; ++v) {
    int k = ((v >> 2) << 4) + (hi << 3) + ((v & 3) << 1);
    f.u[v] = *(const unsigned*)(p + n * stride + k);
  }
  return f.v;
}

// TDM 2D tile load (bf16 elements) per ISA cdna5 §8: D# group0/1 built inline,
// pad feature reproduces the padded LDS row stride used by the fragment readers.
// pad_code: LDS pad of 1 dword inserted every 2^(pad_code+1) dwords.
// Toolchain here exposes the 6-arg builtin: (g0, g1, g2, g3, g4, cpol).
__device__ __forceinline__ void tdm_load_2d_bf16(unsigned lds_off, const void* gptr,
                                                 unsigned tile_d0, unsigned tile_d1,
                                                 unsigned tensor_d0, unsigned tensor_d1,
                                                 unsigned stride0, unsigned pad_code) {
  unsigned long long ga = (unsigned long long)gptr;
  u32x4 g0;
  g0[0] = 1u;                                   // count=1, user descriptor
  g0[1] = lds_off;                              // LDS byte address
  g0[2] = (unsigned)ga;                         // global_addr[31:0]
  g0[3] = (unsigned)(ga >> 32) | (2u << 30);    // global_addr[56:32] | type=2 (image)
  i32x8 g1;
  g1[0] = (int)((1u << 16) | (1u << 20) | (pad_code << 22)); // data_size=2B, pad_enable, pad_amount=1 dword
  g1[1] = (int)(tensor_d0 << 16);                            // atomic_barrier_addr=0 | tensor_dim0 lo
  g1[2] = (int)((tensor_d0 >> 16) | (tensor_d1 << 16));      // tensor_dim0 hi | tensor_dim1 lo
  g1[3] = (int)((tensor_d1 >> 16) | (tile_d0 << 16));        // tensor_dim1 hi | tile_dim0
  g1[4] = (int)(tile_d1 & 0xFFFFu);                          // tile_dim1 | tile_dim2=0
  g1[5] = (int)stride0;                                      // tensor_dim0_stride lo
  g1[6] = 0;                                                 // stride0 hi | stride1 lo
  g1[7] = 0;
  i32x4 g2 = {0, 0, 0, 0};
  i32x4 g3 = {0, 0, 0, 0};
  i32x8 g4 = {0, 0, 0, 0, 0, 0, 0, 0};
  __builtin_amdgcn_tensor_load_to_lds(g0, g1, g2, g3, g4, 0);
}

// MODE 0: fp32 X in, bf16 head-major [B,H,S,DK] out, fused (x+b)*scale
// MODE 1: bf16 X in (workspace, TDM-staged), fp32 flat [N,D] out, fused +b
// Block tile 128x64, 8 waves, each wave a 32x32 tile (4 WMMAs / k-step).
template<int MODE>
__global__ void __launch_bounds__(256)
proj_gemm_kernel(const void* __restrict__ Xv, const float* __restrict__ W,
                 const float* __restrict__ bias, void* __restrict__ Yv, float scale) {
  __shared__ struct {
    unsigned short A[128 * 34];   // 128 rows x 32 (+2 pad) bf16  (offset 0 for TDM)
    unsigned short Bt[64 * 34];
  } sm;

  const int tid  = threadIdx.x;
  const int wave = tid >> 5, lane = tid & 31;
  const int n = lane & 15, hi = lane >> 4;
  const int m0 = blockIdx.y * 128;
  const int n0 = blockIdx.x * 64;
  const int wm = wave >> 1;    // 0..3 : 32-row stripe
  const int wn = wave & 1;     // 0..1 : 32-col half

  v8f acc00 = {}, acc01 = {}, acc10 = {}, acc11 = {};

  const float*          Xf = (const float*)Xv;
  const unsigned short* Xh = (const unsigned short*)Xv;

  const int arow = tid >> 1;           // 0..127
  const int aseg = (tid & 1) << 4;     // 0,16
  const int wrow = tid >> 2;           // 0..63
  const int wseg = (tid & 3) << 3;     // 0,8,16,24

  for (int k0 = 0; k0 < D_; k0 += 32) {
    if (MODE == 0) {
      const float* src = Xf + (size_t)(m0 + arow) * D_ + k0 + aseg;
#pragma unroll
      for (int e = 0; e < 16; ++e) sm.A[arow * 34 + aseg + e] = f2bf(src[e]);
    } else {
      if (wave == 0)   // TDM: 128x32 bf16 tile, row stride 1024, pad -> LDS stride 34
        tdm_load_2d_bf16(0u, Xh + (size_t)m0 * D_ + k0,
                         32u, 128u, 32u, 128u, (unsigned)D_, 3u);
    }
    const float* wsrc = W + (size_t)(n0 + wrow) * D_ + k0 + wseg;   // y = x @ W^T
#pragma unroll
    for (int e = 0; e < 8; ++e) sm.Bt[wrow * 34 + wseg + e] = f2bf(wsrc[e]);
    if (MODE == 1 && wave == 0) __builtin_amdgcn_s_wait_tensorcnt(0);
    __syncthreads();

    v16bf a0 = frag_a(sm.A + (wm * 32) * 34,      34, n, hi);
    v16bf a1 = frag_a(sm.A + (wm * 32 + 16) * 34, 34, n, hi);
    v16bf b0 = frag_b(sm.Bt + (wn * 32) * 34,      34, n, hi);
    v16bf b1 = frag_b(sm.Bt + (wn * 32 + 16) * 34, 34, n, hi);
    acc00 = __builtin_amdgcn_wmma_f32_16x16x32_bf16(false, a0, false, b0, (short)0, acc00, false, false);
    acc01 = __builtin_amdgcn_wmma_f32_16x16x32_bf16(false, a0, false, b1, (short)0, acc01, false, false);
    acc10 = __builtin_amdgcn_wmma_f32_16x16x32_bf16(false, a1, false, b0, (short)0, acc10, false, false);
    acc11 = __builtin_amdgcn_wmma_f32_16x16x32_bf16(false, a1, false, b1, (short)0, acc11, false, false);
    __syncthreads();
  }

  const int j0 = n0 + wn * 32 + n;
  const int j1 = j0 + 16;
  const float bia0 = bias[j0], bia1 = bias[j1];

  if (MODE == 0) {
    unsigned short* Y = (unsigned short*)Yv;
    const int h0 = j0 >> 6, d0 = j0 & 63;
    const int h1 = j1 >> 6, d1 = j1 & 63;
#pragma unroll
    for (int i = 0; i < 8; ++i) {
      int gr0 = m0 + wm * 32 + i + (hi << 3);
      int gr1 = gr0 + 16;
      int bb0 = gr0 >> 11, ss0 = gr0 & (S_ - 1);
      int bb1 = gr1 >> 11, ss1 = gr1 & (S_ - 1);
      Y[((size_t)(bb0 * H_ + h0) * S_ + ss0) * DK_ + d0] = f2bf((acc00[i] + bia0) * scale);
      Y[((size_t)(bb0 * H_ + h1) * S_ + ss0) * DK_ + d1] = f2bf((acc01[i] + bia1) * scale);
      Y[((size_t)(bb1 * H_ + h0) * S_ + ss1) * DK_ + d0] = f2bf((acc10[i] + bia0) * scale);
      Y[((size_t)(bb1 * H_ + h1) * S_ + ss1) * DK_ + d1] = f2bf((acc11[i] + bia1) * scale);
    }
  } else {
    float* Y = (float*)Yv;
#pragma unroll
    for (int i = 0; i < 8; ++i) {
      int gr0 = m0 + wm * 32 + i + (hi << 3);
      int gr1 = gr0 + 16;
      Y[(size_t)gr0 * D_ + j0] = acc00[i] + bia0;
      Y[(size_t)gr0 * D_ + j1] = acc01[i] + bia1;
      Y[(size_t)gr1 * D_ + j0] = acc10[i] + bia0;
      Y[(size_t)gr1 * D_ + j1] = acc11[i] + bia1;
    }
  }
}

// Flash attention: one wave = 16 query rows, 32-key steps, online softmax.
// K tile staged by TDM (tensor_load_to_lds), V staged transposed manually.
__global__ void __launch_bounds__(256)
flash_attn_kernel(const unsigned short* __restrict__ Qb, const unsigned short* __restrict__ Kb,
                  const unsigned short* __restrict__ Vb, const int* __restrict__ mask,
                  unsigned short* __restrict__ Ob) {
  __shared__ struct {
    unsigned short K[32 * 66];      // 32 keys x 64 dk (+2 pad)  (offset 0 for TDM)
    unsigned short V[64 * 34];      // transposed: 64 dk x 32 keys (+2 pad)
    unsigned short P[8][16 * 34];   // per-wave P re-layout scratch
  } sm;

  const int tid  = threadIdx.x;
  const int wave = tid >> 5, lane = tid & 31;
  const int n = lane & 15, hi = lane >> 4;
  const int bh = blockIdx.y;
  const int b  = bh >> 4;
  const int h  = bh & 15;
  const int q0 = blockIdx.x * 128 + wave * 16;
  const size_t headbase = (size_t)bh * S_ * DK_;

  // Q fragments for both DK chunks, loaded straight from global (pre-scaled by 1/sqrt(DK))
  v16bf aq0, aq1;
  {
    FragU f0, f1;
    const unsigned short* qp = Qb + headbase + (size_t)(q0 + n) * DK_;
#pragma unroll
    for (int v = 0; v < 8; ++v) {
      int k = ((v >> 2) << 4) + (hi << 3) + ((v & 3) << 1);
      f0.u[v] = *(const unsigned*)(qp + k);
      f1.u[v] = *(const unsigned*)(qp + 32 + k);
    }
    aq0 = f0.v; aq1 = f1.v;
  }

  v8f o0 = {}, o1 = {}, o2 = {}, o3 = {};
  float rmax[8], rsum[8];
#pragma unroll
  for (int i = 0; i < 8; ++i) { rmax[i] = -1e30f; rsum[i] = 0.f; }

  unsigned short* sPw = &sm.P[wave][0];
  const int* mrow = mask + (size_t)b * S_ * S_;

  for (int kb = 0; kb < S_; kb += 32) {
    // K tile via Tensor Data Mover: 32x64 bf16, stride 64, pad -> LDS stride 66
    if (wave == 0)
      tdm_load_2d_bf16(0u, Kb + headbase + (size_t)kb * DK_,
                       64u, 32u, 64u, 32u, (unsigned)DK_, 4u);
    // V tile staged transposed by all threads
    {
      const unsigned short* vs = Vb + headbase + (size_t)kb * DK_;
#pragma unroll
      for (int x = 0; x < 8; ++x) {
        int e = tid * 8 + x;
        int key = e >> 6, dk = e & 63;
        sm.V[dk * 34 + key] = vs[e];
      }
    }
    if (wave == 0) __builtin_amdgcn_s_wait_tensorcnt(0);
    __syncthreads();

    if (kb + 32 < S_)
      __builtin_prefetch(Vb + headbase + (size_t)(kb + 32) * DK_ + tid * 8, 0, 0);

    // scores: S(16q x 32keys) = Q(16x64) x K^T, two N tiles x two DK chunks
    v8f s0 = {}, s1 = {};
    {
      v16bf bk;
      bk = frag_b(sm.K, 66, n, hi);
      s0 = __builtin_amdgcn_wmma_f32_16x16x32_bf16(false, aq0, false, bk, (short)0, s0, false, false);
      bk = frag_b(sm.K + 32, 66, n, hi);
      s0 = __builtin_amdgcn_wmma_f32_16x16x32_bf16(false, aq1, false, bk, (short)0, s0, false, false);
      bk = frag_b(sm.K + 16 * 66, 66, n, hi);
      s1 = __builtin_amdgcn_wmma_f32_16x16x32_bf16(false, aq0, false, bk, (short)0, s1, false, false);
      bk = frag_b(sm.K + 16 * 66 + 32, 66, n, hi);
      s1 = __builtin_amdgcn_wmma_f32_16x16x32_bf16(false, aq1, false, bk, (short)0, s1, false, false);
    }

    // mask + online softmax (row stats via wave32 16-lane xor reductions)
#pragma unroll
    for (int i = 0; i < 8; ++i) {
      int q = q0 + i + (hi << 3);
      const int* mp = mrow + (size_t)q * S_ + kb;
      if (mp[n] == 0)      s0[i] = -1e9f;
      if (mp[n + 16] == 0) s1[i] = -1e9f;

      float mx = fmaxf(s0[i], s1[i]);
#pragma unroll
      for (int off = 8; off >= 1; off >>= 1)
        mx = fmaxf(mx, __shfl_xor(mx, off, 32));
      float nm = fmaxf(rmax[i], mx);
      float alpha = __expf(rmax[i] - nm);
      rmax[i] = nm;
      float p0 = __expf(s0[i] - nm);
      float p1 = __expf(s1[i] - nm);
      float ps = p0 + p1;
#pragma unroll
      for (int off = 8; off >= 1; off >>= 1)
        ps += __shfl_xor(ps, off, 32);
      rsum[i] = rsum[i] * alpha + ps;
      o0[i] *= alpha; o1[i] *= alpha; o2[i] *= alpha; o3[i] *= alpha;
      int prow = i + (hi << 3);
      sPw[prow * 34 + n]      = f2bf(p0);
      sPw[prow * 34 + n + 16] = f2bf(p1);
    }

    // O(16x64) += P(16x32) x V(32x64): 4 N tiles over DK
    {
      v16bf ap = frag_a(sPw, 34, n, hi);
      v16bf bv;
      bv = frag_b(sm.V,           34, n, hi);
      o0 = __builtin_amdgcn_wmma_f32_16x16x32_bf16(false, ap, false, bv, (short)0, o0, false, false);
      bv = frag_b(sm.V + 16 * 34, 34, n, hi);
      o1 = __builtin_amdgcn_wmma_f32_16x16x32_bf16(false, ap, false, bv, (short)0, o1, false, false);
      bv = frag_b(sm.V + 32 * 34, 34, n, hi);
      o2 = __builtin_amdgcn_wmma_f32_16x16x32_bf16(false, ap, false, bv, (short)0, o2, false, false);
      bv = frag_b(sm.V + 48 * 34, 34, n, hi);
      o3 = __builtin_amdgcn_wmma_f32_16x16x32_bf16(false, ap, false, bv, (short)0, o3, false, false);
    }
    __syncthreads();
  }

  // normalize and emit bf16 [B,S,D] for the output projection
#pragma unroll
  for (int i = 0; i < 8; ++i) {
    int q = q0 + i + (hi << 3);
    float inv = 1.0f / rsum[i];
    size_t base = ((size_t)b * S_ + q) * D_ + (size_t)h * DK_;
    Ob[base + n]      = f2bf(o0[i] * inv);
    Ob[base + 16 + n] = f2bf(o1[i] * inv);
    Ob[base + 32 + n] = f2bf(o2[i] * inv);
    Ob[base + 48 + n] = f2bf(o3[i] * inv);
  }
}

extern "C" void kernel_launch(void* const* d_in, const int* in_sizes, int n_in,
                              void* d_out, int out_size, void* d_ws, size_t ws_size,
                              hipStream_t stream) {
  (void)in_sizes; (void)n_in; (void)out_size; (void)ws_size;
  const float* query = (const float*)d_in[0];
  const float* key   = (const float*)d_in[1];
  const float* value = (const float*)d_in[2];
  const int*   mask  = (const int*)d_in[3];
  const float* Wq = (const float*)d_in[4];
  const float* bq = (const float*)d_in[5];
  const float* Wk = (const float*)d_in[6];
  const float* bk = (const float*)d_in[7];
  const float* Wv = (const float*)d_in[8];
  const float* bv = (const float*)d_in[9];
  const float* Wo = (const float*)d_in[10];
  const float* bo = (const float*)d_in[11];
  float* out = (float*)d_out;

  const size_t elems = (size_t)B_ * S_ * D_;    // 8,388,608 bf16 per tensor
  unsigned short* Qw = (unsigned short*)d_ws;
  unsigned short* Kw = Qw + elems;
  unsigned short* Vw = Kw + elems;
  unsigned short* Aw = Vw + elems;

  dim3 blk(256);
  dim3 gproj(D_ / 64, (B_ * S_) / 128);         // 16 x 64
  proj_gemm_kernel<0><<<gproj, blk, 0, stream>>>((const void*)query, Wq, bq, (void*)Qw, 0.125f);
  proj_gemm_kernel<0><<<gproj, blk, 0, stream>>>((const void*)key,   Wk, bk, (void*)Kw, 1.0f);
  proj_gemm_kernel<0><<<gproj, blk, 0, stream>>>((const void*)value, Wv, bv, (void*)Vw, 1.0f);

  dim3 gattn(S_ / 128, B_ * H_);                // 16 x 64
  flash_attn_kernel<<<gattn, blk, 0, stream>>>(Qw, Kw, Vw, mask, Aw);

  proj_gemm_kernel<1><<<gproj, blk, 0, stream>>>((const void*)Aw, Wo, bo, (void*)out, 1.0f);
}
